// GCNModel3_13924283974407
// MI455X (gfx1250) — matrix operand order repaired and checked by reading, not compiled, not run on hardware
//
#include <hip/hip_runtime.h>
#include <hip/hip_bf16.h>
#include <math.h>

typedef __attribute__((ext_vector_type(16))) _Float16 v16h;
typedef __attribute__((ext_vector_type(8)))  _Float16 v8h;
typedef __attribute__((ext_vector_type(8)))  float    v8f;

#define GN 50000
#define GE 800000
#define GH 256

// ---------------------------------------------------------------------------
// WMMA GEMM: Out[M x Nc] = act( (A .* rowScale?) [M x K] @ W [K x Nc] + bias )
// Block: 256 threads = 8 waves. Tile 64x64, K-step 32.
// Wave (wr in 0..3, wc in 0..1) computes rows wr*16..+15, cols wc*32..+31
// via two v_wmma_f32_16x16x32_f16 per K-step.
// ---------------------------------------------------------------------------
#define BM 64
#define BN 64
#define BK 32
#define LDA 40   // halves per A row (pad, 80B = 16B aligned stride)
#define LDB 40   // halves per transposed-B column

__global__ __launch_bounds__(256)
void gemm_wmma(const float* __restrict__ A, const float* __restrict__ W,
               const float* __restrict__ bias, const float* __restrict__ rowScale,
               float* __restrict__ Out, int M, int Kdim, int Ncols, int relu)
{
    __shared__ _Float16 sA[BM * LDA];
    __shared__ _Float16 sB[BN * LDB];

    const int tid  = threadIdx.x;
    const int lane = tid & 31;
    const int wave = tid >> 5;
    const int wr   = wave & 3;   // row slab 0..3
    const int wc   = wave >> 2;  // col slab 0..1
    const int m0   = blockIdx.x * BM;
    const int n0   = blockIdx.y * BN;

    v8f acc0 = {};
    v8f acc1 = {};

    for (int k0 = 0; k0 < Kdim; k0 += BK) {
        // ---- stage A tile [64 x 32] as f16 (with optional per-row scale) ----
        {
            const int idx = tid * 8;          // 0..2047
            const int r   = idx >> 5;         // row in tile
            const int c   = idx & 31;         // col in tile
            const int gr  = m0 + r;
            v8h v;
            if (gr < M) {
                float scale = rowScale ? rowScale[gr] : 1.0f;
                const float* ap = A + (size_t)gr * Kdim + k0 + c;
                #pragma unroll
                for (int i = 0; i < 8; ++i) v[i] = (_Float16)(ap[i] * scale);
            } else {
                #pragma unroll
                for (int i = 0; i < 8; ++i) v[i] = (_Float16)0.0f;
            }
            *(v8h*)&sA[r * LDA + c] = v;
        }
        // ---- stage B tile [32 x 64] transposed: sB[c*LDB + k] = W[k0+k][n0+c]
        {
            #pragma unroll
            for (int i = 0; i < 8; ++i) {
                const int linear = tid + i * 256;   // 0..2047
                const int k = linear >> 6;
                const int c = linear & 63;
                sB[c * LDB + k] = (_Float16)W[(size_t)(k0 + k) * Ncols + n0 + c];
            }
        }
        __syncthreads();

        // ---- A fragment (contiguous k order, same as B) ----
        // lane L: row = L&15, kbase = 16*(L>=16); halves 0..15 -> K=kbase..+15
        // (two contiguous 16B LDS reads)
        const int r   = (wr << 4) + (lane & 15);
        const int kbA = (lane >= 16) ? 16 : 0;
        const v8h a_lo = *(const v8h*)&sA[r * LDA + kbA];
        const v8h a_hi = *(const v8h*)&sA[r * LDA + kbA + 8];
        v16h afrag;
        #pragma unroll
        for (int i = 0; i < 8; ++i) { afrag[i] = a_lo[i]; afrag[i + 8] = a_hi[i]; }

        // ---- B fragments (contiguous k order, same as A) ----
        // lane L: col = L&15, kbase = 16*(L>=16); halves 0..15 -> K=kbase..+15
        // transposed staging makes this two contiguous 16B LDS reads
        const int kbB = (lane >= 16) ? 16 : 0;
        const int c0  = (wc << 5) + (lane & 15);
        const v8h b0lo = *(const v8h*)&sB[c0 * LDB + kbB];
        const v8h b0hi = *(const v8h*)&sB[c0 * LDB + kbB + 8];
        const int c1  = c0 + 16;
        const v8h b1lo = *(const v8h*)&sB[c1 * LDB + kbB];
        const v8h b1hi = *(const v8h*)&sB[c1 * LDB + kbB + 8];
        v16h bfrag0, bfrag1;
        #pragma unroll
        for (int i = 0; i < 8; ++i) {
            bfrag0[i] = b0lo[i]; bfrag0[i + 8] = b0hi[i];
            bfrag1[i] = b1lo[i]; bfrag1[i + 8] = b1hi[i];
        }

        acc0 = __builtin_amdgcn_wmma_f32_16x16x32_f16(false, afrag, false, bfrag0,
                                                      (short)0, acc0, false, false);
        acc1 = __builtin_amdgcn_wmma_f32_16x16x32_f16(false, afrag, false, bfrag1,
                                                      (short)0, acc1, false, false);
        __syncthreads();
    }

    // ---- writeback (C/D layout: VGPR i -> M = i + 8*(lane>=16), N = lane&15)
    const int colBase = n0 + (wc << 5) + (lane & 15);
    const int rowOff  = (lane >= 16) ? 8 : 0;
    const float b0 = bias[colBase];
    const float b1 = bias[colBase + 16];
    #pragma unroll
    for (int i = 0; i < 8; ++i) {
        const int row = m0 + (wr << 4) + rowOff + i;
        if (row < M) {
            float v0 = acc0[i] + b0;
            float v1 = acc1[i] + b1;
            if (relu) { v0 = fmaxf(v0, 0.0f); v1 = fmaxf(v1, 0.0f); }
            Out[(size_t)row * Ncols + colBase]      = v0;
            Out[(size_t)row * Ncols + colBase + 16] = v1;
        }
    }
}

// ---------------------------------------------------------------------------
// degree accumulation + inverse-sqrt norms
// ---------------------------------------------------------------------------
__global__ void deg_accum(const int* __restrict__ src, const int* __restrict__ dst,
                          float* __restrict__ dout, float* __restrict__ din, int E)
{
    int e = blockIdx.x * blockDim.x + threadIdx.x;
    if (e < E) {
        atomicAdd(&dout[src[e]], 1.0f);
        atomicAdd(&din[dst[e]], 1.0f);
    }
}

__global__ void inv_sqrt_k(float* __restrict__ ns, float* __restrict__ nd, int n)
{
    int i = blockIdx.x * blockDim.x + threadIdx.x;
    if (i < n) {
        float a = ns[i], b = nd[i];
        ns[i] = a > 0.0f ? rsqrtf(a) : 0.0f;
        nd[i] = b > 0.0f ? rsqrtf(b) : 0.0f;
    }
}

// ---------------------------------------------------------------------------
// GCN message: agg[dst] += x[src] * ns[src]   (64 threads x float4 per edge)
// ---------------------------------------------------------------------------
__global__ void gcn_msg(const float* __restrict__ x, const float* __restrict__ ns,
                        const int* __restrict__ src, const int* __restrict__ dst,
                        float* __restrict__ agg, int E)
{
    long long gid = (long long)blockIdx.x * blockDim.x + threadIdx.x;
    long long total = (long long)E * 64;
    if (gid >= total) return;
    int e = (int)(gid >> 6);
    int c = (int)(gid & 63) << 2;
    int s = src[e], d = dst[e];
    float scale = ns[s];
    float4 v = *(const float4*)(x + (size_t)s * GH + c);
    float* ap = agg + (size_t)d * GH + c;
    atomicAdd(ap + 0, v.x * scale);
    atomicAdd(ap + 1, v.y * scale);
    atomicAdd(ap + 2, v.z * scale);
    atomicAdd(ap + 3, v.w * scale);
}

// ---------------------------------------------------------------------------
// attention edges: one wave per (edge, head); DH=64, wave32 xor-reduce
// ---------------------------------------------------------------------------
__global__ __launch_bounds__(256)
void edge_attn(const float* __restrict__ Q, const float* __restrict__ Kf,
               const float* __restrict__ V, const int* __restrict__ src,
               const int* __restrict__ dst, float* __restrict__ wv,
               float* __restrict__ z, int E)
{
    const int lane = threadIdx.x & 31;
    const int wave = threadIdx.x >> 5;
    long long e = (long long)blockIdx.x * 2 + (wave >> 2);
    const int h = wave & 3;
    if (e >= E) return;
    const int s = src[e], d = dst[e];
    const float* kp = Kf + (size_t)s * GH + h * 64;
    const float* qp = Q  + (size_t)d * GH + h * 64;
    float partial = kp[lane] * qp[lane] + kp[lane + 32] * qp[lane + 32];
    #pragma unroll
    for (int off = 16; off > 0; off >>= 1)
        partial += __shfl_xor(partial, off, 32);
    float sc = partial * 0.125f;             // / sqrt(64)
    sc = fminf(fmaxf(sc, -10.0f), 10.0f);
    sc = expf(sc);
    const float* vp = V + (size_t)s * GH + h * 64;
    float* wp = wv + (size_t)d * GH + h * 64;
    atomicAdd(&wp[lane],      vp[lane]      * sc);
    atomicAdd(&wp[lane + 32], vp[lane + 32] * sc);
    if (lane == 0) atomicAdd(&z[(size_t)d * 4 + h], sc);
}

__global__ void attn_norm(const float* __restrict__ wv, const float* __restrict__ z,
                          float* __restrict__ out, int n)
{
    int i = blockIdx.x * blockDim.x + threadIdx.x;
    if (i < n) {
        int node = i >> 8;
        int h = (i >> 6) & 3;
        out[i] = wv[i] / (z[node * 4 + h] + 1e-6f);
    }
}

__global__ void concat3(const float* __restrict__ x1, const float* __restrict__ x2,
                        const float* __restrict__ x3, float* __restrict__ xc, int n)
{
    int i = blockIdx.x * blockDim.x + threadIdx.x;
    if (i < n) {
        int node = i / 768;
        int j = i - node * 768;
        const float* s = (j < 256) ? x1 : ((j < 512) ? x2 : x3);
        xc[i] = s[(size_t)node * GH + (j & 255)];
    }
}

// ---------------------------------------------------------------------------
// final head: sigmoid(h2 @ W3 + b3), one wave per node
// ---------------------------------------------------------------------------
__global__ __launch_bounds__(256)
void head_out(const float* __restrict__ h2, const float* __restrict__ W3,
              const float* __restrict__ b3, float* __restrict__ out, int Nn)
{
    const int lane = threadIdx.x & 31;
    const int wave = threadIdx.x >> 5;
    const int node = blockIdx.x * 8 + wave;
    if (node >= Nn) return;
    const float* hp = h2 + (size_t)node * GH;
    float s = 0.0f;
    #pragma unroll
    for (int i = 0; i < 8; ++i) s += hp[lane + i * 32] * W3[lane + i * 32];
    #pragma unroll
    for (int off = 16; off > 0; off >>= 1) s += __shfl_xor(s, off, 32);
    if (lane == 0) out[node] = 1.0f / (1.0f + expf(-(s + b3[0])));
}

// ---------------------------------------------------------------------------
extern "C" void kernel_launch(void* const* d_in, const int* in_sizes, int n_in,
                              void* d_out, int out_size, void* d_ws, size_t ws_size,
                              hipStream_t stream)
{
    (void)in_sizes; (void)n_in; (void)out_size; (void)ws_size;

    const float* features = (const float*)d_in[0];
    const int*   src      = (const int*)d_in[1];
    const int*   dst      = (const int*)d_in[2];
    // d_in[3] edge_types: unused by reference
    const float* Wm = (const float*)d_in[4];
    const float* bm = (const float*)d_in[5];
    const float* W1 = (const float*)d_in[24];
    const float* b1 = (const float*)d_in[25];
    const float* W2 = (const float*)d_in[26];
    const float* b2 = (const float*)d_in[27];
    const float* W3 = (const float*)d_in[28];
    const float* b3 = (const float*)d_in[29];

    const int N_ = GN, E_ = GE;

    // ---- workspace carve (floats) ----
    float* ws  = (float*)d_ws;
    float* ns   = ws;                 // [N]
    float* nd   = ns + N_;            // [N]
    float* x    = nd + N_;            // [N,256]
    float* agg  = x   + (size_t)N_ * 256;
    float* Qb   = agg + (size_t)N_ * 256;
    float* Kb   = Qb  + (size_t)N_ * 256;
    float* Vb   = Kb  + (size_t)N_ * 256;
    float* wv   = Vb  + (size_t)N_ * 256;
    float* z    = wv  + (size_t)N_ * 256;   // [N,4]
    float* x1   = z   + (size_t)N_ * 4;
    float* x2   = x1  + (size_t)N_ * 256;
    float* x3   = x2  + (size_t)N_ * 256;
    float* xcat = x3  + (size_t)N_ * 256;   // [N,768]
    float* h1   = xcat+ (size_t)N_ * 768;   // [N,512]
    float* h2   = h1  + (size_t)N_ * 512;   // [N,256]

    const dim3 gemm256(782, 4);   // ceil(50000/64) x (256/64)
    const dim3 gemm512(782, 8);

    // ---- degree norms ----
    hipMemsetAsync(ns, 0, (size_t)2 * N_ * sizeof(float), stream);
    deg_accum<<<(E_ + 255) / 256, 256, 0, stream>>>(src, dst, ns, nd, E_);
    inv_sqrt_k<<<(N_ + 255) / 256, 256, 0, stream>>>(ns, nd, N_);

    // ---- x = relu(features @ Wm + bm) ----
    gemm_wmma<<<gemm256, 256, 0, stream>>>(features, Wm, bm, nullptr, x, N_, 256, 256, 1);

    long long msgTot = (long long)E_ * 64;
    unsigned msgBlocks = (unsigned)((msgTot + 255) / 256);

    auto gcn = [&](const float* xin, const float* W, const float* b, float* outb) {
        hipMemsetAsync(agg, 0, (size_t)N_ * 256 * sizeof(float), stream);
        gcn_msg<<<msgBlocks, 256, 0, stream>>>(xin, ns, src, dst, agg, E_);
        gemm_wmma<<<gemm256, 256, 0, stream>>>(agg, W, b, nd, outb, N_, 256, 256, 1);
    };
    auto attn_layer = [&](const float* xin, int wi, float* xout) {
        gcn(xin, (const float*)d_in[wi + 0], (const float*)d_in[wi + 1], Qb);
        gcn(xin, (const float*)d_in[wi + 2], (const float*)d_in[wi + 3], Kb);
        gcn(xin, (const float*)d_in[wi + 4], (const float*)d_in[wi + 5], Vb);
        hipMemsetAsync(wv, 0, (size_t)N_ * 256 * sizeof(float), stream);
        hipMemsetAsync(z,  0, (size_t)N_ * 4   * sizeof(float), stream);
        edge_attn<<<E_ / 2, 256, 0, stream>>>(Qb, Kb, Vb, src, dst, wv, z, E_);
        attn_norm<<<(N_ * 256 + 255) / 256, 256, 0, stream>>>(wv, z, xout, N_ * 256);
    };

    attn_layer(x,  6,  x1);
    attn_layer(x1, 12, x2);
    attn_layer(x2, 18, x3);

    concat3<<<(N_ * 768 + 255) / 256, 256, 0, stream>>>(x1, x2, x3, xcat, N_ * 768);
    gemm_wmma<<<gemm512, 256, 0, stream>>>(xcat, W1, b1, nullptr, h1, N_, 768, 512, 1);
    gemm_wmma<<<gemm256, 256, 0, stream>>>(h1,   W2, b2, nullptr, h2, N_, 512, 256, 1);
    head_out<<<(N_ + 7) / 8, 256, 0, stream>>>(h2, W3, b3, (float*)d_out, N_);
}
